// RegionRelationshipReasoning_28312424415653
// MI455X (gfx1250) — matrix operand order repaired and checked by reading, not compiled, not run on hardware
//
#include <hip/hip_runtime.h>

// Problem constants (from reference)
#define DD    1024          // D
#define DIN   4096
#define NB    64            // batch
#define NR    100           // regions per batch
#define MROWS (NB * NR)     // 6400 flat rows
#define NRP   112           // NR padded to 16
#define KRP   128           // NR padded to 32 (K dim of R@g)

typedef __attribute__((ext_vector_type(16))) __bf16        v16bf;
typedef __attribute__((ext_vector_type(8)))  __bf16        v8bf;
typedef __attribute__((ext_vector_type(8)))  float         v8f;
typedef __attribute__((ext_vector_type(16))) unsigned short v16u;

union CVbf { v16u u; v16bf b; };

__device__ __forceinline__ unsigned short f2bf(float f) {
    union { float f; unsigned u; } v; v.f = f;
    unsigned u = v.u;
    u += 0x7FFFu + ((u >> 16) & 1u);   // round-to-nearest-even
    return (unsigned short)(u >> 16);
}

__device__ __forceinline__ v8f wmma_bf16(v16bf a, v16bf b, v8f c) {
    // (neg_a, A, neg_b, B, c_mod, C, reuse_a, reuse_b)
    return __builtin_amdgcn_wmma_f32_16x16x32_bf16(false, a, false, b, (short)0, c, false, false);
}

// A fragment (16x32 bf16, row-major source, contiguous-K lane layout):
// lane L: row = caller-resolved; K contiguous run at (L/16)*16 .. +15.
__device__ __forceinline__ v16bf load_a_bf(const unsigned short* A, int lda, int mrow, int kb, int lane) {
    int kh = (lane >> 4) * 16;
    const unsigned short* p = A + (size_t)mrow * lda + kb + kh;
    return *(const v16bf*)p;
}

// B fragment for C = A @ W^T (W row-major [N,K]): B[k,n] = W[n,k] so a lane's
// column of B is a contiguous run of W's row: one 32B load per lane.
// Layout: lane L: n = L%16, K halfblock = (L/16)*16, 2 K per VGPR.
__device__ __forceinline__ v16bf load_bT(const unsigned short* W, int ldw, int nrow, int kb, int lane) {
    int kh = (lane >> 4) * 16;
    return *(const v16bf*)(W + (size_t)nrow * ldw + kb + kh);
}

// ---------------- f32 -> bf16 conversion (bandwidth-bound, outside GEMM loops) ----------------
__global__ void cvt_bf16(const float* __restrict__ in, unsigned short* __restrict__ out, int n) {
    int i = blockIdx.x * blockDim.x + threadIdx.x;
    int stride = gridDim.x * blockDim.x;
    for (; i < n; i += stride) out[i] = f2bf(in[i]);
}

// ---------------- Generic flat GEMM: C = A(bf16) @ W^T (+bias +residual) ----------------
// 4x2 16x16 tiles per wave (64x32 macro-tile): 8 WMMAs per 12 b128 loads per k-step.
// ~112 live VGPRs -> no spills (4x4 tiling overflows the 256-VGPR window and spills).
__global__ void gemm_flat_bf16(const unsigned short* __restrict__ A, const unsigned short* __restrict__ W,
                               float* __restrict__ Cf, unsigned short* __restrict__ Cbf,
                               const float* __restrict__ bias, const float* __restrict__ residual,
                               int M, int N, int K) {
    int lane = threadIdx.x & 31;
    int wave = threadIdx.x >> 5;
    int gw = blockIdx.x * (blockDim.x >> 5) + wave;
    int mTiles = M >> 6, nTiles = N >> 5;
    if (gw >= mTiles * nTiles) return;
    int mt = (gw % mTiles) << 6;
    int nt = (gw / mTiles) << 5;

    v8f acc[4][2];
#pragma unroll
    for (int i = 0; i < 4; i++)
#pragma unroll
        for (int j = 0; j < 2; j++) acc[i][j] = (v8f){};

    int am = mt + (lane & 15);
    int bn = nt + (lane & 15);
    for (int kb = 0; kb < K; kb += 32) {
        v16bf fa[4], fb[2];
#pragma unroll
        for (int i = 0; i < 4; i++) fa[i] = load_a_bf(A, K, am + i * 16, kb, lane);
#pragma unroll
        for (int j = 0; j < 2; j++) fb[j] = load_bT(W, K, bn + j * 16, kb, lane);
#pragma unroll
        for (int i = 0; i < 4; i++)
#pragma unroll
            for (int j = 0; j < 2; j++) acc[i][j] = wmma_bf16(fa[i], fb[j], acc[i][j]);
    }

    int cn0 = nt + (lane & 15), cn1 = cn0 + 16;
    int rbase = (lane >> 4) * 8;
    float bz0 = bias ? bias[cn0] : 0.f;
    float bz1 = bias ? bias[cn1] : 0.f;
#pragma unroll
    for (int i = 0; i < 4; i++) {
#pragma unroll
        for (int r = 0; r < 8; r++) {
            int m = mt + i * 16 + rbase + r;
            float v0 = acc[i][0][r] + bz0;
            float v1 = acc[i][1][r] + bz1;
            if (residual) {
                v0 += residual[(size_t)m * N + cn0];
                v1 += residual[(size_t)m * N + cn1];
            }
            if (Cf) {
                Cf[(size_t)m * N + cn0] = v0;
                Cf[(size_t)m * N + cn1] = v1;
            }
            if (Cbf) {
                Cbf[(size_t)m * N + cn0] = f2bf(v0);
                Cbf[(size_t)m * N + cn1] = f2bf(v1);
            }
        }
    }
}

// ---------------- Batched scores: S[b] = phi[b] @ pho[b]^T  (100x100, K=1024) ----------------
__global__ void gemm_scores(const unsigned short* __restrict__ phi, const unsigned short* __restrict__ pho,
                            float* __restrict__ S) {
    int lane = threadIdx.x & 31;
    int wave = threadIdx.x >> 5;
    int gw = blockIdx.x * (blockDim.x >> 5) + wave;
    const int tpb = 7 * 7;                     // 112/16 x 112/16 tiles per batch
    if (gw >= NB * tpb) return;
    int b = gw / tpb;
    int t = gw % tpb;
    int mt = (t % 7) << 4;
    int nt = (t / 7) << 4;
    int amr = mt + (lane & 15); if (amr > NR - 1) amr = NR - 1;   // clamp keeps loads in-bounds
    int bnr = nt + (lane & 15); if (bnr > NR - 1) bnr = NR - 1;
    const unsigned short* Ab = phi + (size_t)b * NR * DD;
    const unsigned short* Wb = pho + (size_t)b * NR * DD;
    v8f acc = {};
    for (int kb = 0; kb < DD; kb += 32) {
        v16bf fa = load_a_bf(Ab, DD, amr, kb, lane);
        v16bf fb = load_bT(Wb, DD, bnr, kb, lane);
        acc = wmma_bf16(fa, fb, acc);
    }
    int n = nt + (lane & 15);
    int rbase = (lane >> 4) * 8;
    float* Sb = S + (size_t)b * NR * NRP;
#pragma unroll
    for (int r = 0; r < 8; r++) {
        int m = mt + rbase + r;
        if (m < NR) Sb[(size_t)m * NRP + n] = acc[r];   // n < 112 always (row stride NRP)
    }
}

// ---------------- Row softmax over m=0..99, write bf16 R rows padded to 128 ----------------
__global__ void softmax_rows(const float* __restrict__ S, unsigned short* __restrict__ R) {
    int lane = threadIdx.x & 31;
    int wave = threadIdx.x >> 5;
    int row = blockIdx.x * (blockDim.x >> 5) + wave;   // 0..6399 = b*100+n
    if (row >= MROWS) return;
    const float* Srow = S + (size_t)row * NRP;
    float x[4], e[4];
#pragma unroll
    for (int i = 0; i < 4; i++) { int m = lane + i * 32; x[i] = (m < NR) ? Srow[m] : -3.0e38f; }
    float mx = fmaxf(fmaxf(x[0], x[1]), fmaxf(x[2], x[3]));
#pragma unroll
    for (int off = 16; off >= 1; off >>= 1) mx = fmaxf(mx, __shfl_xor(mx, off, 32));
    float s = 0.f;
#pragma unroll
    for (int i = 0; i < 4; i++) { int m = lane + i * 32; e[i] = (m < NR) ? __expf(x[i] - mx) : 0.f; s += e[i]; }
#pragma unroll
    for (int off = 16; off >= 1; off >>= 1) s += __shfl_xor(s, off, 32);
    float inv = 1.0f / s;
    unsigned short* Rrow = R + (size_t)row * KRP;
#pragma unroll
    for (int i = 0; i < 4; i++) { int m = lane + i * 32; Rrow[m] = f2bf(e[i] * inv); }  // zeros for m>=100
}

// ---------------- Batched attend: rvwg[b] = R[b] @ g[b]  (K=128, B row-major -> gather) ----------------
__global__ void gemm_attn_apply(const unsigned short* __restrict__ R, const unsigned short* __restrict__ g,
                                unsigned short* __restrict__ rv) {
    int lane = threadIdx.x & 31;
    int wave = threadIdx.x >> 5;
    int gw = blockIdx.x * (blockDim.x >> 5) + wave;
    const int mT = 7, nT = DD / 16;
    if (gw >= NB * mT * nT) return;
    int b = gw / (mT * nT);
    int t = gw % (mT * nT);
    int mt = (t % mT) << 4;
    int nt = (t / mT) << 4;
    int amr = mt + (lane & 15); if (amr > NR - 1) amr = NR - 1;
    const unsigned short* Ab = R + (size_t)b * NR * KRP;
    int n = nt + (lane & 15);
    int kh = (lane >> 4) * 16;
    v8f acc = {};
    for (int kb = 0; kb < KRP; kb += 32) {
        v16bf fa = load_a_bf(Ab, KRP, amr, kb, lane);
        v16u tb;
#pragma unroll
        for (int i = 0; i < 16; i++) {
            int krow = kb + kh + i;
            int grow = b * NR + (krow < NR ? krow : 0);  // pad rows masked by R==0
            tb[i] = g[(size_t)grow * DD + n];
        }
        CVbf cv; cv.u = tb;
        acc = wmma_bf16(fa, cv.b, acc);
    }
    int rbase = (lane >> 4) * 8;
#pragma unroll
    for (int r = 0; r < 8; r++) {
        int m = mt + rbase + r;
        if (m < NR) rv[((size_t)(b * NR + m)) * DD + n] = f2bf(acc[r]);
    }
}

extern "C" void kernel_launch(void* const* d_in, const int* in_sizes, int n_in,
                              void* d_out, int out_size, void* d_ws, size_t ws_size,
                              hipStream_t stream) {
    (void)in_sizes; (void)n_in; (void)out_size; (void)ws_size;
    const float* V      = (const float*)d_in[0];
    const float* W_proj = (const float*)d_in[1];
    const float* W_phi  = (const float*)d_in[2];
    const float* W_pho  = (const float*)d_in[3];
    const float* W_g    = (const float*)d_in[4];
    const float* W_r    = (const float*)d_in[5];
    const float* b_r    = (const float*)d_in[6];
    float* out = (float*)d_out;

    char* w = (char*)d_ws;
    auto carve = [&](size_t bytes) { void* p = (void*)w; w += (bytes + 255) & ~(size_t)255; return p; };
    unsigned short* V_bf     = (unsigned short*)carve((size_t)MROWS * DIN * 2);
    unsigned short* Wproj_bf = (unsigned short*)carve((size_t)DD * DIN * 2);
    unsigned short* Wphi_bf  = (unsigned short*)carve((size_t)DD * DD * 2);
    unsigned short* Wpho_bf  = (unsigned short*)carve((size_t)DD * DD * 2);
    unsigned short* Wg_bf    = (unsigned short*)carve((size_t)DD * DD * 2);
    unsigned short* Wr_bf    = (unsigned short*)carve((size_t)DD * DD * 2);
    float*          Vp_f     = (float*)carve((size_t)MROWS * DD * 4);
    unsigned short* Vp_bf    = (unsigned short*)carve((size_t)MROWS * DD * 2);
    unsigned short* phi_bf   = (unsigned short*)carve((size_t)MROWS * DD * 2);
    unsigned short* pho_bf   = (unsigned short*)carve((size_t)MROWS * DD * 2);
    unsigned short* g_bf     = (unsigned short*)carve((size_t)MROWS * DD * 2);
    float*          S        = (float*)carve((size_t)NB * NR * NRP * 4);
    unsigned short* R_bf     = (unsigned short*)carve((size_t)MROWS * KRP * 2);
    unsigned short* rv_bf    = (unsigned short*)carve((size_t)MROWS * DD * 2);

    const int CT = 256;
    // Conversions (bandwidth-bound; keeps all GEMM inner loops pure bf16 load + wmma)
    cvt_bf16<<<4096, CT, 0, stream>>>(V, V_bf, MROWS * DIN);
    cvt_bf16<<<2048, CT, 0, stream>>>(W_proj, Wproj_bf, DD * DIN);
    cvt_bf16<<<1024, CT, 0, stream>>>(W_phi, Wphi_bf, DD * DD);
    cvt_bf16<<<1024, CT, 0, stream>>>(W_pho, Wpho_bf, DD * DD);
    cvt_bf16<<<1024, CT, 0, stream>>>(W_g, Wg_bf, DD * DD);
    cvt_bf16<<<1024, CT, 0, stream>>>(W_r, Wr_bf, DD * DD);

    int flatTiles = (MROWS / 64) * (DD / 32);   // 3200 wave-tiles -> 400 blocks of 8 waves
    int flatBlocks = (flatTiles + 7) / 8;
    // Vp = V @ W_proj^T  (dominant GEMM, K=4096), write f32 (residual) + bf16 (next GEMMs)
    gemm_flat_bf16<<<flatBlocks, 256, 0, stream>>>(V_bf, Wproj_bf, Vp_f, Vp_bf, nullptr, nullptr, MROWS, DD, DIN);
    gemm_flat_bf16<<<flatBlocks, 256, 0, stream>>>(Vp_bf, Wphi_bf, nullptr, phi_bf, nullptr, nullptr, MROWS, DD, DD);
    gemm_flat_bf16<<<flatBlocks, 256, 0, stream>>>(Vp_bf, Wpho_bf, nullptr, pho_bf, nullptr, nullptr, MROWS, DD, DD);
    gemm_flat_bf16<<<flatBlocks, 256, 0, stream>>>(Vp_bf, Wg_bf, nullptr, g_bf, nullptr, nullptr, MROWS, DD, DD);

    int sTiles = NB * 49;
    gemm_scores<<<(sTiles + 7) / 8, 256, 0, stream>>>(phi_bf, pho_bf, S);
    softmax_rows<<<(MROWS + 7) / 8, 256, 0, stream>>>(S, R_bf);

    int rTiles = NB * 7 * (DD / 16);
    gemm_attn_apply<<<(rTiles + 7) / 8, 256, 0, stream>>>(R_bf, g_bf, rv_bf);

    gemm_flat_bf16<<<flatBlocks, 256, 0, stream>>>(rv_bf, Wr_bf, out, nullptr, b_r, Vp_f, MROWS, DD, DD);
}
